// QuantLinear_24077586661404
// MI455X (gfx1250) — compile-verified, matrix operand order checked by reading, not run on hardware
//
#include <hip/hip_runtime.h>

typedef _Float16 half4 __attribute__((ext_vector_type(4)));
typedef _Float16 half8 __attribute__((ext_vector_type(8)));
typedef _Float16 v16h  __attribute__((ext_vector_type(16)));
typedef float    v8f   __attribute__((ext_vector_type(8)));
typedef int      v4i   __attribute__((ext_vector_type(4)));

#define B_   32
#define K_   4096
#define N_   14336
#define QBLK 64
#define PADK (K_ + 8)              // halves per LDS row (bank-conflict padding)
#define LDS_BYTES (B_ * PADK * 2)  // 262,656 B <= 320 KB WGP LDS

__global__ __launch_bounds__(256)
void qlinear_wmma_f16(const float* __restrict__ x,
                      const int* __restrict__ qw,
                      const _Float16* __restrict__ scales,
                      const float* __restrict__ bias,
                      float* __restrict__ out)
{
    extern __shared__ _Float16 xs[];   // [B_][PADK] f16 copy of x

    const int tid = threadIdx.x;

    // ---- Stage x: fp32 global -> f16 LDS (once per workgroup) ----
    {
        const float4* xv = (const float4*)x;
        for (int idx = tid; idx < (B_ * K_) / 4; idx += 256) {
            float4 v = xv[idx];               // global_load_b128
            int flat = idx << 2;
            int row  = flat >> 12;            // / 4096
            int col  = flat & (K_ - 1);
            half4 h;
            h[0] = (_Float16)v.x; h[1] = (_Float16)v.y;
            h[2] = (_Float16)v.z; h[3] = (_Float16)v.w;
            *(half4*)(xs + row * PADK + col) = h;   // ds_store_b64
        }
    }
    __syncthreads();

    const int wave = tid >> 5;
    const int lane = tid & 31;
    const int nt   = lane & 15;        // output column within tile / B-matrix column
    const int hi   = lane >> 4;        // half-wave select
    const int koff = hi << 3;          // K offset +8 for lanes 16..31

    const int n = blockIdx.x * 128 + wave * 16 + nt;

    const int*      wrow = qw     + (long long)n * K_;
    const _Float16* srow = scales + n * (K_ / QBLK);

    const _Float16* a0base = xs + nt * PADK;          // batch rows 0..15
    const _Float16* a1base = xs + (nt + 16) * PADK;   // batch rows 16..31

    v8f acc0 = {};
    v8f acc1 = {};

    for (int kb = 0; kb < K_; kb += QBLK) {
        const float s = (float)srow[kb >> 6];   // one scale per 64-wide block

#pragma unroll
        for (int step = 0; step < 2; ++step) {
            const int kk = kb + step * 32 + koff;

            // ---- B tile: 16 int32 weights for column n, dequant -> f16 ----
            v4i q0 = *(const v4i*)(wrow + kk);        // K kk+0..3
            v4i q1 = *(const v4i*)(wrow + kk + 4);    // K kk+4..7
            v4i q2 = *(const v4i*)(wrow + kk + 16);   // K kk+16..19
            v4i q3 = *(const v4i*)(wrow + kk + 20);   // K kk+20..23

            v16h bm;
#pragma unroll
            for (int i = 0; i < 4; ++i) {
                bm[i]      = (_Float16)((float)q0[i] * s);
                bm[4 + i]  = (_Float16)((float)q1[i] * s);
                bm[8 + i]  = (_Float16)((float)q2[i] * s);
                bm[12 + i] = (_Float16)((float)q3[i] * s);
            }

            // ---- A tiles from LDS (pre-converted f16) ----
            half8 a0lo = *(const half8*)(a0base + kk);        // ds_load_b128
            half8 a0hi = *(const half8*)(a0base + kk + 16);
            half8 a1lo = *(const half8*)(a1base + kk);
            half8 a1hi = *(const half8*)(a1base + kk + 16);

            v16h a0, a1;
#pragma unroll
            for (int i = 0; i < 8; ++i) {
                a0[i] = a0lo[i]; a0[8 + i] = a0hi[i];
                a1[i] = a1lo[i]; a1[8 + i] = a1hi[i];
            }

            acc0 = __builtin_amdgcn_wmma_f32_16x16x32_f16(
                false, a0, false, bm, (short)0, acc0, false, false);
            acc1 = __builtin_amdgcn_wmma_f32_16x16x32_f16(
                false, a1, false, bm, (short)0, acc1, false, false);
        }
    }

    // ---- Epilogue: bias + store. D layout: VGPR r -> M = r + 8*hi ----
    const float bn = bias[n];
#pragma unroll
    for (int r = 0; r < 8; ++r) {
        const int m0 = r + (hi << 3);
        out[(long long)m0 * N_ + n]        = acc0[r] + bn;
        out[(long long)(m0 + 16) * N_ + n] = acc1[r] + bn;
    }
}

extern "C" void kernel_launch(void* const* d_in, const int* in_sizes, int n_in,
                              void* d_out, int out_size, void* d_ws, size_t ws_size,
                              hipStream_t stream) {
    const float*    x      = (const float*)d_in[0];
    const int*      qw     = (const int*)d_in[1];
    const _Float16* scales = (const _Float16*)d_in[2];
    const float*    bias   = (const float*)d_in[3];
    float*          out    = (float*)d_out;

    dim3 grid(N_ / 128);   // 112 blocks, 8 waves each, 16 N-columns per wave
    dim3 block(256);
    qlinear_wmma_f16<<<grid, block, LDS_BYTES, stream>>>(x, qw, scales, bias, out);
}
